// GraphTransformerModel_80238579023948
// MI455X (gfx1250) — matrix operand order repaired and checked by reading, not compiled, not run on hardware
//
#include <hip/hip_runtime.h>

// ---------------------------------------------------------------------------
// Types for CDNA5 WMMA (wave32): v_wmma_f32_16x16x32_bf16
// ---------------------------------------------------------------------------
typedef __attribute__((ext_vector_type(16))) __bf16 v16bf;
typedef __attribute__((ext_vector_type(8)))  __bf16 v8bf;
typedef __attribute__((ext_vector_type(8)))  float  v8f;

#define HEADS 8

// ---------------------------------------------------------------------------
// Small utility kernels
// ---------------------------------------------------------------------------
__global__ void fill_f32_k(float* __restrict__ p, float v, int count) {
    int i = blockIdx.x * blockDim.x + threadIdx.x;
    if (i < count) p[i] = v;
}

__device__ __forceinline__ void atomicMaxF(float* addr, float val) {
    if (val >= 0.0f) atomicMax((int*)addr, __float_as_int(val));
    else             atomicMin((unsigned int*)addr, __float_as_uint(val));
}

// ---------------------------------------------------------------------------
// Weight pack: W[k x m] f32 row-major -> bf16 WMMA B-fragment layout.
// Output index: (((ct * (k/32) + kt) * 32 + lane) * 16 + j)
//   lane: col = ct*16 + (lane&15), khalf = lane>>4
//   slot j: K = kt*32 + khalf*8 + (j&7) + (j>>3)*16
// In the GEMM each lane reads its 16-element fragment as one aligned 32B load.
// ---------------------------------------------------------------------------
__global__ void pack_w_k(const float* __restrict__ W, __bf16* __restrict__ Wp,
                         int k, int m) {
    int i = blockIdx.x * blockDim.x + threadIdx.x;
    if (i >= k * m) return;
    int j    = i & 15;
    int lane = (i >> 4) & 31;
    int tile = i >> 9;
    int KT   = k >> 5;
    int kt   = tile % KT;
    int ct   = tile / KT;
    int col  = ct * 16 + (lane & 15);
    int krow = kt * 32 + (lane >> 4) * 8 + (j & 7) + (j >> 3) * 16;
    Wp[i] = (__bf16)W[(size_t)krow * m + col];
}

// ---------------------------------------------------------------------------
// WMMA GEMM: Y[n x m] = X[n x k] @ W[k x m] + bias[m]
//   X : f32 row-major (converted to bf16 while staging through LDS)
//   Wp: bf16, pre-packed fragment layout (see pack_w_k)
//   Y : f32 or bf16 (template)
// Block = 256 threads = 8 waves; grid.x = n/16 row tiles; grid.y = 128-col
// strips. Wave w owns the 16x16 tile at column tile (blockIdx.y*8 + w).
// The whole 16 x k A-strip (k<=256 -> <=8KB) is staged ONCE into LDS, already
// permuted into the ISA 16-bit A 16x32 fragment order, so the inner loop is
// just: 2x ds_load_b128 (A) + 2x global_load_b128 (B, L2-hot) + 1x WMMA.
// Requires n % 16 == 0, k % 32 == 0, k <= 256.
// ---------------------------------------------------------------------------
template <bool BF16OUT>
__global__ void wmma_gemm_k(const float* __restrict__ X,
                            const __bf16* __restrict__ Wp,
                            const float* __restrict__ bias,
                            void* __restrict__ Y,
                            int n, int k, int m) {
    __shared__ alignas(32) __bf16 As[8 * 512];  // up to k=256: (k/32) tiles * 512

    const int tid  = threadIdx.x;
    const int wave = tid >> 5;
    const int lane = tid & 31;
    const int row0 = blockIdx.x * 16;
    const int KT   = k >> 5;                    // # of 32-wide K tiles
    const int colTile = blockIdx.y * 8 + wave;  // global 16-col tile index
    const bool active = (colTile * 16 < m);

    // ---- stage entire A strip, shuffled into fragment order ----
    // element (kt, r, c): goes to lane (r + 16*((c>>3)&1)), slot j where
    // pair<4: c = 2p + 8*khalf + off ; pair>=4: c = 16 + 2(p-4) + 8*khalf + off
    for (int i = tid; i < KT * 512; i += 256) {
        int kt = i >> 9;
        int r  = (i >> 5) & 15;
        int c  = i & 31;
        int off   = c & 1;
        int khalf = (c >> 3) & 1;
        int c2    = c - khalf * 8 - off;
        int pair  = (c2 < 8) ? (c2 >> 1) : (4 + ((c2 - 16) >> 1));
        int j     = pair * 2 + off;
        int dstLane = r + (khalf << 4);
        As[(kt << 9) + dstLane * 16 + j] =
            (__bf16)X[(size_t)(row0 + r) * k + (kt * 32 + c)];
    }
    __syncthreads();

    if (active) {
        v8f acc = {};
        const __bf16* wpBase = Wp + ((size_t)colTile * KT) * 512;
        #pragma unroll 4
        for (int kt = 0; kt < KT; ++kt) {
            v16bf a = *(const v16bf*)&As[(kt << 9) + (lane << 4)];
            v16bf b = *(const v16bf*)(wpBase + (size_t)kt * 512 + (lane << 4));
            acc = __builtin_amdgcn_wmma_f32_16x16x32_bf16(
                false, a, false, b, (short)0, acc, false, false);
        }
        // C/D 16x16 f32 layout: lane<16 -> N=lane, rows 0..7; lane>=16 -> rows 8..15
        const int cn    = lane & 15;
        const int rbase = (lane >> 4) * 8;
        const int gcol  = colTile * 16 + cn;
        const float bv  = bias[gcol];
        #pragma unroll
        for (int r = 0; r < 8; ++r) {
            int grow = row0 + rbase + r;
            float val = acc[r] + bv;
            if (BF16OUT) ((__bf16*)Y)[(size_t)grow * m + gcol] = (__bf16)val;
            else         ((float*)Y)[(size_t)grow * m + gcol] = val;
        }
    }
}

// ---------------------------------------------------------------------------
// Edge kernels (bandwidth-dominant gather/scatter; q/k/v stored bf16 and read
// as 16-byte vectors -> global_load_b128). One thread per (edge, head).
// ---------------------------------------------------------------------------
__global__ void edge_score_k(const __bf16* __restrict__ q, const __bf16* __restrict__ kbuf,
                             const int* __restrict__ src, const int* __restrict__ dst,
                             float* __restrict__ s, int nedge, int d, int stride, float scale) {
    int idx = blockIdx.x * blockDim.x + threadIdx.x;
    if (idx >= nedge * HEADS) return;
    int e = idx >> 3, h = idx & 7;
    const __bf16* qp = q    + (size_t)src[e] * stride + h * d;
    const __bf16* kp = kbuf + (size_t)dst[e] * stride + h * d;
    float acc = 0.0f;
    for (int i = 0; i < d; i += 8) {
        v8bf qv = *(const v8bf*)(qp + i);
        v8bf kv = *(const v8bf*)(kp + i);
        #pragma unroll
        for (int j = 0; j < 8; ++j) acc += (float)qv[j] * (float)kv[j];
    }
    s[idx] = acc * scale;
}

__global__ void edge_max_k(const float* __restrict__ s, const int* __restrict__ dst,
                           float* __restrict__ m, int nedge) {
    int idx = blockIdx.x * blockDim.x + threadIdx.x;
    if (idx >= nedge * HEADS) return;
    int e = idx >> 3, h = idx & 7;
    atomicMaxF(&m[(size_t)dst[e] * HEADS + h], s[idx]);
}

__global__ void edge_exp_k(float* __restrict__ s, const int* __restrict__ dst,
                           const float* __restrict__ m, float* __restrict__ z, int nedge) {
    int idx = blockIdx.x * blockDim.x + threadIdx.x;
    if (idx >= nedge * HEADS) return;
    int e = idx >> 3, h = idx & 7;
    size_t mi = (size_t)dst[e] * HEADS + h;
    float ex = __expf(s[idx] - m[mi]);
    s[idx] = ex;
    atomicAdd(&z[mi], ex);
}

// att[dst*attStride + headMul*h*d + c] += (e/z) * outScale * v[src,h,c]
// Layer 0: attStride=128, headMul=1, outScale=1   (per-head concat)
// Layer 1: attStride=64,  headMul=0, outScale=1/8 (mean over heads folded in)
__global__ void edge_agg_k(const __bf16* __restrict__ v, const float* __restrict__ ebuf,
                           const float* __restrict__ z, const int* __restrict__ src,
                           const int* __restrict__ dst, float* __restrict__ att,
                           int nedge, int d, int stride, int attStride, int headMul,
                           float outScale) {
    int idx = blockIdx.x * blockDim.x + threadIdx.x;
    if (idx >= nedge * HEADS) return;
    int e = idx >> 3, h = idx & 7;
    int dn = dst[e];
    float p = ebuf[idx] / z[(size_t)dn * HEADS + h] * outScale;
    const __bf16* vp = v + (size_t)src[e] * stride + h * d;
    float* op = att + (size_t)dn * attStride + headMul * h * d;
    for (int i = 0; i < d; i += 8) {
        v8bf vv = *(const v8bf*)(vp + i);
        #pragma unroll
        for (int j = 0; j < 8; ++j) atomicAdd(&op[i + j], p * (float)vv[j]);
    }
}

// ---------------------------------------------------------------------------
// Fused gated-residual node kernels
// ---------------------------------------------------------------------------
template <int NT>
__device__ __forceinline__ float block_sum(float v, float* red) {
    int t = threadIdx.x;
    red[t] = v; __syncthreads();
    #pragma unroll
    for (int s = NT / 2; s > 0; s >>= 1) {
        if (t < s) red[t] += red[t + s];
        __syncthreads();
    }
    float r = red[0];
    __syncthreads();
    return r;
}

// Layer 0: gate + layernorm + relu, H=128, one block (128 thr) per node
__global__ void gated_ln_relu_128_k(const float* __restrict__ xa, const float* __restrict__ ra,
                                    const float* __restrict__ gW, const float* __restrict__ lng,
                                    const float* __restrict__ lnb, float* __restrict__ out) {
    __shared__ float red[128];
    size_t n = blockIdx.x; int t = threadIdx.x;
    float x = xa[n * 128 + t];
    float r = ra[n * 128 + t];
    float gd = block_sum<128>(x * gW[t] + r * gW[128 + t] + (x - r) * gW[256 + t], red);
    float g = 1.0f / (1.0f + __expf(-gd));
    float o = x * g + r * (1.0f - g);
    float mu  = block_sum<128>(o, red) * (1.0f / 128.0f);
    float var = block_sum<128>((o - mu) * (o - mu), red) * (1.0f / 128.0f);
    float y = (o - mu) * rsqrtf(var + 1e-5f) * lng[t] + lnb[t];
    out[n * 128 + t] = fmaxf(y, 0.0f);
}

// Layer 1: gate only (only_gate=True), C=64, one block (64 thr) per node
__global__ void gated_only_64_k(const float* __restrict__ xa, const float* __restrict__ ra,
                                const float* __restrict__ gW, float* __restrict__ out) {
    __shared__ float red[64];
    size_t n = blockIdx.x; int t = threadIdx.x;
    float x = xa[n * 64 + t];
    float r = ra[n * 64 + t];
    float gd = block_sum<64>(x * gW[t] + r * gW[64 + t] + (x - r) * gW[128 + t], red);
    float g = 1.0f / (1.0f + __expf(-gd));
    out[n * 64 + t] = x * g + r * (1.0f - g);
}

// ---------------------------------------------------------------------------
// Host-side orchestration
// ---------------------------------------------------------------------------
extern "C" void kernel_launch(void* const* d_in, const int* in_sizes, int n_in,
                              void* d_out, int out_size, void* d_ws, size_t ws_size,
                              hipStream_t stream) {
    const int N = 100000, E = 1600000, F = 256, H = 128, C = 64;

    const float* x       = (const float*)d_in[0];
    const int*   src     = (const int*)  d_in[1];
    const int*   dst     = (const int*)  d_in[2];
    const float* in_W    = (const float*)d_in[3];
    const float* in_b    = (const float*)d_in[4];
    const float* q0_W    = (const float*)d_in[5];
    const float* q0_b    = (const float*)d_in[6];
    const float* k0_W    = (const float*)d_in[7];
    const float* k0_b    = (const float*)d_in[8];
    const float* v0_W    = (const float*)d_in[9];
    const float* v0_b    = (const float*)d_in[10];
    const float* res0_W  = (const float*)d_in[11];
    const float* res0_b  = (const float*)d_in[12];
    const float* gate0_W = (const float*)d_in[13];
    const float* ln0_g   = (const float*)d_in[14];
    const float* ln0_b   = (const float*)d_in[15];
    const float* q1_W    = (const float*)d_in[16];
    const float* q1_b    = (const float*)d_in[17];
    const float* k1_W    = (const float*)d_in[18];
    const float* k1_b    = (const float*)d_in[19];
    const float* v1_W    = (const float*)d_in[20];
    const float* v1_b    = (const float*)d_in[21];
    const float* res1_W  = (const float*)d_in[22];
    const float* res1_b  = (const float*)d_in[23];
    const float* gate1_W = (const float*)d_in[24];

    // ---- workspace carve-up (all offsets 256B aligned) ----
    char* ws = (char*)d_ws;
    size_t off = 0;
    auto take = [&](size_t bytes) -> void* {
        void* p = ws + off;
        off = (off + bytes + 255) & ~(size_t)255;
        return p;
    };

    __bf16* wb   = (__bf16*)take((size_t)303104 * 2);  // packed bf16 weight pool
    __bf16* w_in = wb;                 // 256*128 = 32768
    __bf16* w_q0 = w_in + 32768;       // 128*128 = 16384
    __bf16* w_k0 = w_q0 + 16384;
    __bf16* w_v0 = w_k0 + 16384;
    __bf16* w_r0 = w_v0 + 16384;
    __bf16* w_q1 = w_r0 + 16384;       // 128*512 = 65536
    __bf16* w_k1 = w_q1 + 65536;
    __bf16* w_v1 = w_k1 + 65536;
    __bf16* w_r1 = w_v1 + 65536;       // 128*64 = 8192

    float*  buf_h   = (float*) take((size_t)N * H * 4);    // node features
    float*  buf_res = (float*) take((size_t)N * H * 4);    // residual projection
    float*  buf_att = (float*) take((size_t)N * H * 4);    // attention accumulator
    __bf16* buf_q   = (__bf16*)take((size_t)N * 512 * 2);  // q (bf16)
    __bf16* buf_k   = (__bf16*)take((size_t)N * 512 * 2);  // k (bf16)
    __bf16* buf_v   = (__bf16*)take((size_t)N * 512 * 2);  // v (bf16)
    float*  buf_s   = (float*) take((size_t)E * HEADS * 4);// scores / exp
    float*  buf_m   = (float*) take((size_t)N * HEADS * 4);// segment max
    float*  buf_z   = (float*) take((size_t)N * HEADS * 4);// segment sum

    auto pack = [&](const float* s, __bf16* d, int kk, int mm) {
        pack_w_k<<<(kk * mm + 255) / 256, 256, 0, stream>>>(s, d, kk, mm);
    };
    auto fill = [&](float* p, float v, int cnt) {
        fill_f32_k<<<(cnt + 255) / 256, 256, 0, stream>>>(p, v, cnt);
    };

    // ---- weights f32 -> packed bf16 fragments (once per launch) ----
    pack(in_W,   w_in, F, H);
    pack(q0_W,   w_q0, H, H);
    pack(k0_W,   w_k0, H, H);
    pack(v0_W,   w_v0, H, H);
    pack(res0_W, w_r0, H, H);
    pack(q1_W,   w_q1, H, 512);
    pack(k1_W,   w_k1, H, 512);
    pack(v1_W,   w_v1, H, 512);
    pack(res1_W, w_r1, H, C);

    const int rowTiles = N / 16;           // 6250
    const int edgeThr  = E * HEADS;        // 12.8M
    const int edgeBlk  = (edgeThr + 255) / 256;

    // ---- h = x @ in_W + in_b ----
    wmma_gemm_k<false><<<dim3(rowTiles, 1), 256, 0, stream>>>(x, w_in, in_b, buf_h, N, F, H);

    // ---- layer 0: q/k/v (bf16 out) ----
    wmma_gemm_k<true><<<dim3(rowTiles, 1), 256, 0, stream>>>(buf_h, w_q0, q0_b, buf_q, N, H, H);
    wmma_gemm_k<true><<<dim3(rowTiles, 1), 256, 0, stream>>>(buf_h, w_k0, k0_b, buf_k, N, H, H);
    wmma_gemm_k<true><<<dim3(rowTiles, 1), 256, 0, stream>>>(buf_h, w_v0, v0_b, buf_v, N, H, H);

    // ---- layer 0 edge softmax + aggregate (d=16, stride=H) ----
    edge_score_k<<<edgeBlk, 256, 0, stream>>>(buf_q, buf_k, src, dst, buf_s, E, 16, H, 0.25f);
    fill(buf_m, -3.0e38f, N * HEADS);
    fill(buf_z, 0.0f, N * HEADS);
    fill(buf_att, 0.0f, N * H);
    edge_max_k<<<edgeBlk, 256, 0, stream>>>(buf_s, dst, buf_m, E);
    edge_exp_k<<<edgeBlk, 256, 0, stream>>>(buf_s, dst, buf_m, buf_z, E);
    edge_agg_k<<<edgeBlk, 256, 0, stream>>>(buf_v, buf_s, buf_z, src, dst, buf_att,
                                            E, 16, H, H, 1, 1.0f);

    // ---- layer 0 gated residual + LN + relu -> new h ----
    wmma_gemm_k<false><<<dim3(rowTiles, 1), 256, 0, stream>>>(buf_h, w_r0, res0_b, buf_res, N, H, H);
    gated_ln_relu_128_k<<<N, 128, 0, stream>>>(buf_att, buf_res, gate0_W, ln0_g, ln0_b, buf_h);

    // ---- layer 1: q/k/v (128 -> 512, bf16 out) ----
    wmma_gemm_k<true><<<dim3(rowTiles, 4), 256, 0, stream>>>(buf_h, w_q1, q1_b, buf_q, N, H, 512);
    wmma_gemm_k<true><<<dim3(rowTiles, 4), 256, 0, stream>>>(buf_h, w_k1, k1_b, buf_k, N, H, 512);
    wmma_gemm_k<true><<<dim3(rowTiles, 4), 256, 0, stream>>>(buf_h, w_v1, v1_b, buf_v, N, H, 512);

    // ---- layer 1 edge softmax + aggregate (d=64, stride=512, head-mean folded) ----
    edge_score_k<<<edgeBlk, 256, 0, stream>>>(buf_q, buf_k, src, dst, buf_s, E, 64, 512, 0.125f);
    fill(buf_m, -3.0e38f, N * HEADS);
    fill(buf_z, 0.0f, N * HEADS);
    fill(buf_att, 0.0f, N * C);
    edge_max_k<<<edgeBlk, 256, 0, stream>>>(buf_s, dst, buf_m, E);
    edge_exp_k<<<edgeBlk, 256, 0, stream>>>(buf_s, dst, buf_m, buf_z, E);
    edge_agg_k<<<edgeBlk, 256, 0, stream>>>(buf_v, buf_s, buf_z, src, dst, buf_att,
                                            E, 64, 512, C, 0, 0.125f);

    // ---- layer 1 gated residual (only_gate) -> d_out ----
    wmma_gemm_k<false><<<dim3(rowTiles, 1), 256, 0, stream>>>(buf_h, w_r1, res1_b, buf_res, N, H, C);
    gated_only_64_k<<<N, 64, 0, stream>>>(buf_att, buf_res, gate1_W, (float*)d_out);
}
